// VisualConceptInterface_70068096467254
// MI455X (gfx1250) — matrix operand order repaired and checked by reading, not compiled, not run on hardware
//
#include <hip/hip_runtime.h>
#include <hip/hip_bf16.h>

// ---------------------------------------------------------------------------
// Shapes (fixed by the reference)
// ---------------------------------------------------------------------------
#define Bsz 4096
#define Dd  1024
#define Cc  100
#define Hh  512
#define R1n 50
#define R2n 25

#define BROWS 64             // batch rows per block (4 row-subtiles x 16)
#define BCOLS 64             // staged W1 columns per pass (4 h-chunks)
#define ASTRIDE 1032         // 1024 + 8 bf16 pad (2064B row -> 4-bank skew)
#define BSTRIDE 1032

typedef __attribute__((ext_vector_type(16))) __bf16 bf16x16;
typedef __attribute__((ext_vector_type(8)))  __bf16 bf16x8;
typedef __attribute__((ext_vector_type(8)))  float  f32x8;

__device__ __forceinline__ __bf16 f2bf(float f) {
    unsigned u = __builtin_bit_cast(unsigned, f);
    u += 0x7FFFu + ((u >> 16) & 1u);          // round-to-nearest-even
    unsigned short s = (unsigned short)(u >> 16);
    return __builtin_bit_cast(__bf16, s);
}

// A frag (16x32 bf16): lane row fixed; Ks = [k0+8*hf,+8) U [k0+16+8*hf,+8)
__device__ __forceinline__ bf16x16 ldA(const __bf16* arow, int k0, int hf) {
    bf16x8 lo = *(const bf16x8*)(arow + k0 + 8 * hf);
    bf16x8 hi = *(const bf16x8*)(arow + k0 + 16 + 8 * hf);
    return __builtin_shufflevector(lo, hi, 0,1,2,3,4,5,6,7,8,9,10,11,12,13,14,15);
}
// B frag (32x16 bf16): lane col fixed; Ks = [k0+8*hf,+8) U [k0+16+8*hf,+8)
__device__ __forceinline__ bf16x16 ldB(const __bf16* bcol, int k0, int hf) {
    bf16x8 lo = *(const bf16x8*)(bcol + k0 + 8 * hf);
    bf16x8 hi = *(const bf16x8*)(bcol + k0 + 16 + 8 * hf);
    return __builtin_shufflevector(lo, hi, 0,1,2,3,4,5,6,7,8,9,10,11,12,13,14,15);
}

// ---------------------------------------------------------------------------
// Kernel 1: fused 100-way detector  acts = sigmoid(relu(g@W1[c]+b1[c])@W2[c]+b2[c])
// Block = (concept c, 64 batch rows). 8 waves = 4 row-subtiles x 2 col-groups.
// Each wave holds 2 accumulators sharing one A fragment (3 LDS loads / wmma),
// with explicit register double-buffering so LDS loads overlap the wmmas.
// ---------------------------------------------------------------------------
__global__ __launch_bounds__(256)
void detector_kernel(const float* __restrict__ feat,   // [B, D]
                     const float* __restrict__ W1,     // [C, D, H]
                     const float* __restrict__ b1,     // [C, H]
                     const float* __restrict__ W2,     // [C, H]
                     const float* __restrict__ b2,     // [C]
                     float* __restrict__ acts)         // [B, C]
{
    extern __shared__ char smem[];
    __bf16* As     = (__bf16*)smem;                                   // [64][ASTRIDE]
    __bf16* Bs     = (__bf16*)(smem + BROWS * ASTRIDE * 2);           // [64][BSTRIDE] (col-major: Bs[n][k])
    float*  actsum = (float*)(smem + (BROWS * ASTRIDE + BCOLS * BSTRIDE) * 2); // [2][64]

    const int c     = blockIdx.x;              // 0..99
    const int bbase = blockIdx.y * BROWS;
    const int tid   = threadIdx.x;             // 0..255
    const int wave  = tid >> 5;                // 0..7
    const int lane  = tid & 31;
    const int hf    = lane >> 4;               // half: 0 or 1
    const int la    = lane & 15;               // A-row / B-col / D-col index
    const int rsub  = wave & 3;                // row subtile: rows [rsub*16, +16)
    const int cpair = wave >> 2;               // column-pair group: staged cols [cpair*32, +32)

    // ---- stage A: 64x1024 f32 -> bf16 in LDS (float4 loads) ----
    for (int i = tid; i < BROWS * (Dd / 4); i += 256) {
        const int r = i >> 8;                  // 256 float4 per row
        const int k = (i & 255) * 4;
        const float4 v = *(const float4*)(feat + ((size_t)(bbase + r) << 10) + k);
        __bf16* dst = As + r * ASTRIDE + k;
        dst[0] = f2bf(v.x); dst[1] = f2bf(v.y); dst[2] = f2bf(v.z); dst[3] = f2bf(v.w);
    }

    const __bf16* arow  = As + (rsub * 16 + la) * ASTRIDE;            // this lane's A row
    const __bf16* bcol0 = Bs + (cpair * 32 + la) * BSTRIDE;           // acc0 column
    const __bf16* bcol1 = bcol0 + 16 * BSTRIDE;                       // acc1 column

    float psum[8];
#pragma unroll
    for (int r = 0; r < 8; ++r) psum[r] = 0.f;

    for (int pass = 0; pass < Hh / BCOLS; ++pass) {        // 8 passes of 64 h-cols
        const int hb = pass * BCOLS;

        __syncthreads();   // previous Bs consumers done (and A staged, 1st iter)
        // ---- stage B chunk: W1[c, 0:1024, hb:hb+64] -> Bs[n][k] bf16 ----
        for (int i = tid; i < Dd * (BCOLS / 4); i += 256) {   // 16384 float4 tasks
            const int k  = i >> 4;
            const int nb = (i & 15) * 4;
            const float4 v = *(const float4*)(W1 + ((size_t)c * Dd + k) * Hh + hb + nb);
            Bs[(nb + 0) * BSTRIDE + k] = f2bf(v.x);
            Bs[(nb + 1) * BSTRIDE + k] = f2bf(v.y);
            Bs[(nb + 2) * BSTRIDE + k] = f2bf(v.z);
            Bs[(nb + 3) * BSTRIDE + k] = f2bf(v.w);
        }
        __syncthreads();

        // ---- two 16x16 tiles over K=1024, shared A frag, SW-pipelined ----
        f32x8 acc0 = {};
        f32x8 acc1 = {};
        bf16x16 a_c  = ldA(arow,  0, hf);
        bf16x16 b0_c = ldB(bcol0, 0, hf);
        bf16x16 b1_c = ldB(bcol1, 0, hf);
#pragma unroll 2
        for (int kb = 0; kb < Dd / 32 - 1; ++kb) {
            const int k1 = (kb + 1) * 32;
            // issue next iteration's fragment loads first (stay in flight
            // while this iteration's wmmas execute)
            bf16x16 a_n  = ldA(arow,  k1, hf);
            bf16x16 b0_n = ldB(bcol0, k1, hf);
            bf16x16 b1_n = ldB(bcol1, k1, hf);
            acc0 = __builtin_amdgcn_wmma_f32_16x16x32_bf16(
                       false, a_c, false, b0_c, (short)0, acc0, false, false);
            acc1 = __builtin_amdgcn_wmma_f32_16x16x32_bf16(
                       false, a_c, false, b1_c, (short)0, acc1, false, false);
            a_c = a_n; b0_c = b0_n; b1_c = b1_n;
        }
        acc0 = __builtin_amdgcn_wmma_f32_16x16x32_bf16(
                   false, a_c, false, b0_c, (short)0, acc0, false, false);
        acc1 = __builtin_amdgcn_wmma_f32_16x16x32_bf16(
                   false, a_c, false, b1_c, (short)0, acc1, false, false);

        // ---- fused epilogue: relu(+b1) then dot with W2 column slices ----
        const int   n0   = hb + cpair * 32 + la;        // global h index, acc0
        const int   n1   = n0 + 16;                     // global h index, acc1
        const float b1v0 = b1[(size_t)c * Hh + n0];
        const float w2v0 = W2[(size_t)c * Hh + n0];
        const float b1v1 = b1[(size_t)c * Hh + n1];
        const float w2v1 = W2[(size_t)c * Hh + n1];
#pragma unroll
        for (int r = 0; r < 8; ++r) {
            float h0 = fmaxf(acc0[r] + b1v0, 0.f);
            float h1 = fmaxf(acc1[r] + b1v1, 0.f);
            psum[r] = fmaf(h0, w2v0, psum[r]);
            psum[r] = fmaf(h1, w2v1, psum[r]);
        }
    }

    // ---- reduce psum over the 16 N-lanes of each half-wave ----
#pragma unroll
    for (int r = 0; r < 8; ++r) {
        psum[r] += __shfl_xor(psum[r], 1, 32);
        psum[r] += __shfl_xor(psum[r], 2, 32);
        psum[r] += __shfl_xor(psum[r], 4, 32);
        psum[r] += __shfl_xor(psum[r], 8, 32);
    }
    if (la == 0) {
        const int rbase = rsub * 16 + hf * 8;           // unique rows per (rsub, hf)
#pragma unroll
        for (int r = 0; r < 8; ++r) actsum[cpair * BROWS + rbase + r] = psum[r];
    }
    __syncthreads();

    if (tid < BROWS) {
        const float a = actsum[tid] + actsum[BROWS + tid] + b2[c];
        acts[(size_t)(bbase + tid) * Cc + c] = 1.f / (1.f + expf(-a));
    }
}

// ---------------------------------------------------------------------------
// Kernel 2: relations MLP + per-row statistics. One wave per batch row.
// ---------------------------------------------------------------------------
__global__ __launch_bounds__(256)
void post_kernel(const float* __restrict__ acts,  // [B, C]
                 const float* __restrict__ rW1,   // [C, R1]
                 const float* __restrict__ rb1,   // [R1]
                 const float* __restrict__ rW2,   // [R1, R2]
                 const float* __restrict__ rb2,   // [R2]
                 float* __restrict__ rel,         // [B, R2]
                 float* __restrict__ strength,    // [B]
                 float* __restrict__ selectivity, // [B]
                 float* __restrict__ active,      // [B]
                 float* __restrict__ diversity)   // [B]
{
    __shared__ float sAct[8][Cc];
    __shared__ float sH[8][R1n];

    const int wave = threadIdx.x >> 5;
    const int lane = threadIdx.x & 31;
    const int b    = blockIdx.x * 8 + wave;

    // load this row's 100 activations; per-lane partial stats
    float sum = 0.f, mx = -1e30f, c05 = 0.f, c01 = 0.f;
#pragma unroll
    for (int q = 0; q < 4; ++q) {
        const int cc = lane + 32 * q;
        if (cc < Cc) {
            const float a = acts[(size_t)b * Cc + cc];
            sAct[wave][cc] = a;
            sum += a;
            mx = fmaxf(mx, a);
            c05 += (a > 0.5f) ? 1.f : 0.f;
            c01 += (a > 0.1f) ? 1.f : 0.f;
        }
    }
#pragma unroll
    for (int m = 16; m >= 1; m >>= 1) {
        sum += __shfl_xor(sum, m, 32);
        mx   = fmaxf(mx, __shfl_xor(mx, m, 32));
        c05 += __shfl_xor(c05, m, 32);
        c01 += __shfl_xor(c01, m, 32);
    }
    if (lane == 0) {
        const float mean = sum * (1.f / Cc);
        strength[b]    = mean;
        selectivity[b] = mx / (mean + 1e-8f);
        active[b]      = c05;
        diversity[b]   = c01 * (1.f / Cc);
    }
    __syncthreads();

    // hidden: 50 units; lane handles j=lane and (lane+32 if <50)
    float h1 = rb1[lane];
    float h2 = (lane < R1n - 32) ? rb1[lane + 32] : 0.f;
    for (int cc = 0; cc < Cc; ++cc) {
        const float av = sAct[wave][cc];
        h1 = fmaf(av, rW1[cc * R1n + lane], h1);
        if (lane < R1n - 32) h2 = fmaf(av, rW1[cc * R1n + lane + 32], h2);
    }
    sH[wave][lane] = fmaxf(h1, 0.f);
    if (lane < R1n - 32) sH[wave][lane + 32] = fmaxf(h2, 0.f);
    __syncthreads();

    if (lane < R2n) {
        float o = rb2[lane];
        for (int j = 0; j < R1n; ++j) o = fmaf(sH[wave][j], rW2[j * R2n + lane], o);
        rel[(size_t)b * R2n + lane] = o;
    }
}

// ---------------------------------------------------------------------------
// Launch
// ---------------------------------------------------------------------------
extern "C" void kernel_launch(void* const* d_in, const int* in_sizes, int n_in,
                              void* d_out, int out_size, void* d_ws, size_t ws_size,
                              hipStream_t stream) {
    const float* feat = (const float*)d_in[0];
    const float* W1   = (const float*)d_in[1];
    const float* b1   = (const float*)d_in[2];
    const float* W2   = (const float*)d_in[3];
    const float* b2   = (const float*)d_in[4];
    const float* rW1  = (const float*)d_in[5];
    const float* rb1  = (const float*)d_in[6];
    const float* rW2  = (const float*)d_in[7];
    const float* rb2  = (const float*)d_in[8];

    float* out         = (float*)d_out;
    float* acts        = out;                                   // 409600
    float* rel         = out + 409600;                          // 102400
    float* strength    = out + 512000;                          // 4096
    float* selectivity = out + 516096;                          // 4096
    float* integrated  = out + 520192;                          // 4194304
    float* active      = out + 4714496;                         // 4096
    float* diversity   = out + 4718592;                         // 4096

    const size_t smem = (size_t)(BROWS * ASTRIDE + BCOLS * BSTRIDE) * 2 + 2 * BROWS * 4;

    dim3 g1(Cc, Bsz / BROWS);          // 100 x 64 blocks
    detector_kernel<<<g1, 256, smem, stream>>>(feat, W1, b1, W2, b2, acts);

    post_kernel<<<Bsz / 8, 256, 0, stream>>>(acts, rW1, rb1, rW2, rb2,
                                             rel, strength, selectivity, active, diversity);

    hipMemcpyAsync(integrated, feat, (size_t)Bsz * Dd * sizeof(float),
                   hipMemcpyDeviceToDevice, stream);
}
